// BinaryLinear_78108275245420
// MI455X (gfx1250) — 8-bit matrix operand order repaired and checked by reading, compiled, not run on hardware
//
#include <hip/hip_runtime.h>
#include <stdint.h>

typedef __attribute__((ext_vector_type(8))) int v8i;
typedef __attribute__((ext_vector_type(4))) unsigned int v4u;
typedef __attribute__((ext_vector_type(4))) int v4i;

#define N_ROWS 8192
#define K_DIM  2048
#define O_DIM  2048
#define BLK_M  256
#define BLK_N  128
#define KC     64
#define LDS_PITCH 80

__global__ __launch_bounds__(256) void sign_quant_kernel(
    const float4* __restrict__ in, int* __restrict__ out, int n4) {
  int idx = blockIdx.x * blockDim.x + threadIdx.x;
  if (idx >= n4) return;
  float4 v = in[idx];
  int s0 = (v.x > 0.f) - (v.x < 0.f);
  int s1 = (v.y > 0.f) - (v.y < 0.f);
  int s2 = (v.z > 0.f) - (v.z < 0.f);
  int s3 = (v.w > 0.f) - (v.w < 0.f);
  out[idx] = (s0 & 0xff) | ((s1 & 0xff) << 8) | ((s2 & 0xff) << 16) | (s3 << 24);
}

__device__ __forceinline__ void tdm_load_tile(uint32_t lds_addr,
                                              const int8_t* gptr,
                                              uint32_t rows) {
  uint64_t ga = (uint64_t)(uintptr_t)gptr;
  v4u g0;
  g0[0] = 1u;
  g0[1] = lds_addr;
  g0[2] = (uint32_t)ga;
  g0[3] = (uint32_t)((ga >> 32) & 0x01FFFFFFu)
          | (2u << 30);
  v8i g1;
  g1[0] = (int)((1u << 20)
              | (3u << 22)
              | (3u << 25));
  g1[1] = (int)((uint32_t)KC << 16);
  g1[2] = (int)(rows << 16);
  g1[3] = (int)((uint32_t)KC << 16);
  g1[4] = (int)rows;
  g1[5] = (int)K_DIM;
  g1[6] = 0;
  g1[7] = 0;
  v4i z4 = {0, 0, 0, 0};
  v8i z8 = {0, 0, 0, 0, 0, 0, 0, 0};
  __builtin_amdgcn_tensor_load_to_lds(g0, g1, z4, z4, z8,  0);
}

__global__ __launch_bounds__(256) void bin_gemm_iu8_tdm_kernel(
    const int8_t* __restrict__ sx, const int8_t* __restrict__ sw,
    const float* __restrict__ alpha, float* __restrict__ out) {
  __shared__ __align__(16) int8_t shA[2][BLK_M * LDS_PITCH];
  __shared__ __align__(16) int8_t shB[2][BLK_N * LDS_PITCH];

  const int lane   = threadIdx.x & 31;
  const int wave   = threadIdx.x >> 5;
  const int laneLo = lane & 15;
  const int laneHi = lane >> 4;
  const int wm = (wave >> 1) * 64;
  const int wn = (wave & 1) * 64;

  const int8_t* gA = sx + (size_t)(blockIdx.y * BLK_M) * K_DIM;
  const int8_t* gB = sw + (size_t)(blockIdx.x * BLK_N) * K_DIM;

  if (threadIdx.x < 32) {
    tdm_load_tile((uint32_t)(uintptr_t)&shA[0][0], gA, BLK_M);
    tdm_load_tile((uint32_t)(uintptr_t)&shB[0][0], gB, BLK_N);
    __builtin_amdgcn_s_wait_tensorcnt(0);
  }
  __syncthreads();

  v8i acc[4][4] = {};

  const int KSTEPS = K_DIM / KC;
  for (int ks = 0; ks < KSTEPS; ++ks) {
    const int cur = ks & 1;
    const int nxt = cur ^ 1;

    if (ks + 1 < KSTEPS && threadIdx.x < 32) {
      const int kn = (ks + 1) * KC;
      tdm_load_tile((uint32_t)(uintptr_t)&shA[nxt][0], gA + kn, BLK_M);
      tdm_load_tile((uint32_t)(uintptr_t)&shB[nxt][0], gB + kn, BLK_N);
    }

    const int8_t* As = &shA[cur][0];
    const int8_t* Bs = &shB[cur][0];

    v8i a[4], b[4];
#pragma unroll
    for (int i = 0; i < 4; ++i) {
      const int8_t* p = As + (wm + i * 16 + laneLo) * LDS_PITCH + laneHi * 16;
      int2 d0 = *(const int2*)(p);
      int2 d1 = *(const int2*)(p + 8);
      int2 d2 = *(const int2*)(p + 32);
      int2 d3 = *(const int2*)(p + 40);
      a[i] = (v8i){d0.x, d0.y, d1.x, d1.y, d2.x, d2.y, d3.x, d3.y};
    }
#pragma unroll
    for (int j = 0; j < 4; ++j) {
      const int8_t* p = Bs + (wn + j * 16 + laneLo) * LDS_PITCH + laneHi * 16;
      int4 q0 = *(const int4*)(p);
      int4 q1 = *(const int4*)(p + 32);
      b[j] = (v8i){q0.x, q0.y, q0.z, q0.w, q1.x, q1.y, q1.z, q1.w};
    }

#pragma unroll
    for (int i = 0; i < 4; ++i)
#pragma unroll
      for (int j = 0; j < 4; ++j)
        acc[i][j] = __builtin_amdgcn_wmma_i32_16x16x64_iu8(
             true, a[i],  true, b[j], acc[i][j],
             false,  false);

    if (ks + 1 < KSTEPS) {
      if (threadIdx.x < 32) __builtin_amdgcn_s_wait_tensorcnt(0);
      __syncthreads();
    }
  }

  const float alp = alpha[0];
#pragma unroll
  for (int i = 0; i < 4; ++i) {
#pragma unroll
    for (int j = 0; j < 4; ++j) {
      const int col = blockIdx.x * BLK_N + wn + j * 16 + laneLo;
#pragma unroll
      for (int r = 0; r < 8; ++r) {
        const int row = blockIdx.y * BLK_M + wm + i * 16 + laneHi * 8 + r;
        out[(size_t)row * O_DIM + col] = alp * (float)acc[i][j][r];
      }
    }
  }
}

extern "C" void kernel_launch(void* const* d_in, const int* in_sizes, int n_in,
                              void* d_out, int out_size, void* d_ws, size_t ws_size,
                              hipStream_t stream) {
  const float* x     = (const float*)d_in[0];
  const float* w     = (const float*)d_in[1];
  const float* alpha = (const float*)d_in[2];
  float* out = (float*)d_out;

  int8_t* sx = (int8_t*)d_ws;
  int8_t* sw = sx + (size_t)N_ROWS * K_DIM;

  const int n4x = (N_ROWS * K_DIM) / 4;
  const int n4w = (O_DIM * K_DIM) / 4;
  sign_quant_kernel<<<(n4x + 255) / 256, 256, 0, stream>>>(
      (const float4*)x, (int*)sx, n4x);
  sign_quant_kernel<<<(n4w + 255) / 256, 256, 0, stream>>>(
      (const float4*)w, (int*)sw, n4w);

  dim3 grid(O_DIM / BLK_N, N_ROWS / BLK_M);
  bin_gemm_iu8_tdm_kernel<<<grid, 256, 0, stream>>>(sx, sw, alpha, out);
}
